// DiffAttn_35794257445301
// MI455X (gfx1250) — compile-verified, matrix operand order checked by reading, not run on hardware
//
#include <hip/hip_runtime.h>
#include <math.h>

// ---------------- problem constants ----------------
#define BATCH 4
#define SEQ   2048
#define DEMB  1024
#define E2    2048            // 2*DEMB
#define LAMBDA_INIT 0.8f
#define EPSV 1e-5f

typedef __attribute__((ext_vector_type(16))) __bf16 v16bf;
typedef __attribute__((ext_vector_type(8)))  float  v8f;

union FragU { unsigned u[8]; v16bf v; };

// round-to-nearest-even f32 -> bf16 (as raw u16)
__device__ __forceinline__ unsigned short f32_to_bf16(float x) {
  unsigned u = __float_as_uint(x);
  unsigned r = u + 0x7FFFu + ((u >> 16) & 1u);
  return (unsigned short)(r >> 16);
}

// K-offset pattern for 16-bit A/B fragments (CDNA5 ISA 7.12.2):
// vgpr g<4: K = (hi?8:0)+2g ; g>=4: K = (hi?24:16)+2(g-4); dword = pair {K,K+1}
__device__ __forceinline__ int kbase_of(int g, int hi) {
  return (g < 4) ? ((hi ? 8 : 0) + 2 * g) : ((hi ? 24 : 16) + 2 * (g - 4));
}

// Inverse mapping: element (r, k) of a 16x32 16-bit operand lives in
// lane = 16*hi + r, packed element e, with:
__device__ __forceinline__ int frag_slot(int r, int k) {
  int hi = (k >> 3) & 1;
  int e  = (k & 7) | ((k >> 1) & 8);
  return ((hi << 4) + r) * 16 + e;   // ushort index inside a 512-entry buffer
}

// Fragment from LDS staged in fragment order: lane reads 16 contiguous bf16
// (two ds_load_b128).
__device__ __forceinline__ v16bf load_frag_lds(const unsigned short* p, int lane) {
  FragU f;
  const unsigned* q = (const unsigned*)(p + (lane << 4));
#pragma unroll
  for (int g = 0; g < 8; ++g) f.u[g] = q[g];
  return f.v;
}

// Fragment straight from global memory, 16 bf16 from one row, K = 8hi..8hi+7 and 16+8hi..23+8hi
// (used by the QK^T kernel where both operands are contiguous along d).
__device__ __forceinline__ v16bf load_frag_rowK(const unsigned short* p, int hi) {
  const unsigned* p32 = (const unsigned*)p;
  FragU f;
#pragma unroll
  for (int g = 0; g < 8; ++g) f.u[g] = p32[kbase_of(g, hi) >> 1];
  return f.v;
}

// ---------------- conversion ----------------
__global__ void __launch_bounds__(256) cvt_f32_bf16(const float* __restrict__ src,
                                                    unsigned short* __restrict__ dst, int n) {
  int i = blockIdx.x * 256 + threadIdx.x;
  if (i < n) dst[i] = f32_to_bf16(src[i]);
}

// ---------------- lambda reduction ----------------
__global__ void __launch_bounds__(256) lambda_kernel(const float* __restrict__ lq1,
                                                     const float* __restrict__ lk1,
                                                     const float* __restrict__ lq2,
                                                     const float* __restrict__ lk2,
                                                     float* __restrict__ lam_out) {
  __shared__ float r1[256], r2[256];
  int t = threadIdx.x;
  float a = 0.f, b = 0.f;
  for (int i = t; i < DEMB; i += 256) { a += lq1[i] * lk1[i]; b += lq2[i] * lk2[i]; }
  r1[t] = a; r2[t] = b; __syncthreads();
  for (int s = 128; s > 0; s >>= 1) {
    if (t < s) { r1[t] += r1[t + s]; r2[t] += r2[t + s]; }
    __syncthreads();
  }
  if (t == 0) *lam_out = expf(r1[0]) - expf(r2[0]) + LAMBDA_INIT;
}

// ---------------- LDS-staged bf16 WMMA GEMM ----------------
// C[M,N] = A[M,K] (row-major bf16) @ B[K,N] (row-major bf16)
// block = 256 threads (8 waves), tile 64(M) x 64(N); wave (mw,nw) owns two
// 16x16 tiles (M-subtiles 2mw, 2mw+1) x (N-subtile nw). A/B staged in LDS in
// WMMA fragment order -> inner loop is pure ds_load_b128 + v_wmma.
// grid = (N/64, M/64, batches)
__global__ void __launch_bounds__(256)
gemm_bf16_wmma(const unsigned short* __restrict__ A,
               const unsigned short* __restrict__ B,
               float* __restrict__ Cf, unsigned short* __restrict__ Cb,
               int M, int N, int K,
               long long sA, long long sB, long long sC) {
  __shared__ __align__(16) unsigned short fragA[4 * 512];   // 4 M-subtiles
  __shared__ __align__(16) unsigned short fragB[4 * 512];   // 4 N-subtiles
  const int tid  = threadIdx.x;
  const int lane = tid & 31;
  const int wave = tid >> 5;
  const int nw = wave & 3, mw = wave >> 2;
  const int m0 = blockIdx.y * 64;
  const int n0 = blockIdx.x * 64;
  const unsigned short* Ab = A + blockIdx.z * sA;
  const unsigned short* Bb = B + blockIdx.z * sB;
  v8f acc0 = {}, acc1 = {};

  for (int k0 = 0; k0 < K; k0 += 32) {
    // stage A tile 64x32 into fragment order
#pragma unroll
    for (int i = 0; i < 8; ++i) {
      int idx = tid + i * 256;                 // 0..2047
      int r = idx >> 5, k = idx & 31;          // coalesced along k
      const unsigned short* src = &Ab[(long long)(m0 + r) * K + k0 + k];
      if (i == 0 && k0 + 32 < K) __builtin_prefetch(src + 32, 0, 0); // global_prefetch_b8
      fragA[(r >> 4) * 512 + frag_slot(r & 15, k)] = *src;
    }
    // stage B tile 32x64 (transposed) into fragment order
#pragma unroll
    for (int i = 0; i < 8; ++i) {
      int idx = tid + i * 256;
      int k = idx >> 6, n = idx & 63;          // coalesced along n
      fragB[(n >> 4) * 512 + frag_slot(n & 15, k)] =
          Bb[(long long)(k0 + k) * N + n0 + n];
    }
    __syncthreads();
    v16bf bf = load_frag_lds(fragB + nw * 512, lane);
    v16bf a0 = load_frag_lds(fragA + (mw * 2 + 0) * 512, lane);
    v16bf a1 = load_frag_lds(fragA + (mw * 2 + 1) * 512, lane);
    acc0 = __builtin_amdgcn_wmma_f32_16x16x32_bf16(false, a0, false, bf,
                                                   (short)0, acc0, false, false);
    acc1 = __builtin_amdgcn_wmma_f32_16x16x32_bf16(false, a1, false, bf,
                                                   (short)0, acc1, false, false);
    __syncthreads();
  }

  const int lr = lane & 15, hi = lane >> 4;
  const long long cbase = blockIdx.z * sC;
  const int mrow0 = m0 + mw * 32;
  const int col   = n0 + nw * 16 + lr;
#pragma unroll
  for (int r = 0; r < 8; ++r) {
    long long ci0 = cbase + (long long)(mrow0 + r + 8 * hi) * N + col;
    long long ci1 = cbase + (long long)(mrow0 + 16 + r + 8 * hi) * N + col;
    if (Cf) { Cf[ci0] = acc0[r];              Cf[ci1] = acc1[r]; }
    else    { Cb[ci0] = f32_to_bf16(acc0[r]); Cb[ci1] = f32_to_bf16(acc1[r]); }
  }
}

// ---------------- fused dual-softmax differential scores ----------------
// one block = (batch, 16-query tile); 8 waves split 128 key tiles.
// dynamic LDS: s1[16][2048] f32, s2[16][2048] f32, red[640] f32
__global__ void __launch_bounds__(256)
diffattn_scores(const unsigned short* __restrict__ q12,
                const unsigned short* __restrict__ k12,
                const float* __restrict__ lam_ptr,
                unsigned short* __restrict__ Pd) {
  extern __shared__ float smem[];
  float* s1  = smem;                 // 16*2048
  float* s2  = smem + 16 * 2048;     // 16*2048
  float* red = smem + 32 * 2048;     // 640 floats scratch
  const int tid  = threadIdx.x;
  const int lane = tid & 31, wave = tid >> 5;
  const int b  = blockIdx.y;
  const int q0 = blockIdx.x * 16;
  const int lr = lane & 15, hi = lane >> 4;
  const float scale = 0.17677669529663687f;   // 1024^(-1/4)
  const long long qbase = ((long long)b * SEQ + q0 + lr) * E2;

  for (int kt = wave; kt < SEQ / 16; kt += 8) {
    const long long kbase = ((long long)b * SEQ + kt * 16 + lr) * E2;
    v8f a1 = {}, a2 = {};
    for (int d0 = 0; d0 < DEMB; d0 += 32) {
      // q·k^T : both operands contiguous along d (K^T columns = K rows)
      v16bf qa = load_frag_rowK(q12 + qbase + d0, hi);
      v16bf kb = load_frag_rowK(k12 + kbase + d0, hi);
      a1 = __builtin_amdgcn_wmma_f32_16x16x32_bf16(false, qa, false, kb,
                                                   (short)0, a1, false, false);
      v16bf qa2 = load_frag_rowK(q12 + qbase + DEMB + d0, hi);
      v16bf kb2 = load_frag_rowK(k12 + kbase + DEMB + d0, hi);
      a2 = __builtin_amdgcn_wmma_f32_16x16x32_bf16(false, qa2, false, kb2,
                                                   (short)0, a2, false, false);
    }
#pragma unroll
    for (int r = 0; r < 8; ++r) {
      int row = r + 8 * hi;
      s1[row * SEQ + kt * 16 + lr] = a1[r] * scale;
      s2[row * SEQ + kt * 16 + lr] = a2[r] * scale;
    }
  }
  __syncthreads();

  // cooperative dual softmax: 16 threads per query row
  const int row = tid >> 4;    // 0..15
  const int sl  = tid & 15;    // 0..15
  float m1 = -3.0e38f, m2 = -3.0e38f;
  for (int j = sl; j < SEQ; j += 16) {
    m1 = fmaxf(m1, s1[row * SEQ + j]);
    m2 = fmaxf(m2, s2[row * SEQ + j]);
  }
  red[tid] = m1; red[256 + tid] = m2;
  __syncthreads();
  if (sl == 0) {
    float a = red[tid], b2 = red[256 + tid];
    for (int t = 1; t < 16; ++t) { a = fmaxf(a, red[tid + t]); b2 = fmaxf(b2, red[256 + tid + t]); }
    red[512 + row] = a; red[528 + row] = b2;
  }
  __syncthreads();
  m1 = red[512 + row]; m2 = red[528 + row];
  float sum1 = 0.f, sum2 = 0.f;
  for (int j = sl; j < SEQ; j += 16) {
    sum1 += expf(s1[row * SEQ + j] - m1);
    sum2 += expf(s2[row * SEQ + j] - m2);
  }
  __syncthreads();
  red[tid] = sum1; red[256 + tid] = sum2;
  __syncthreads();
  if (sl == 0) {
    float a = 0.f, b2 = 0.f;
    for (int t = 0; t < 16; ++t) { a += red[tid + t]; b2 += red[256 + tid + t]; }
    red[544 + row] = a; red[560 + row] = b2;
  }
  __syncthreads();
  sum1 = red[544 + row]; sum2 = red[560 + row];
  const float lam = *lam_ptr;
  const float i1 = 1.0f / sum1, i2 = lam / sum2;
  unsigned short* prow = Pd + ((long long)b * SEQ + q0 + row) * SEQ;
  for (int j = sl; j < SEQ; j += 16) {
    float p = expf(s1[row * SEQ + j] - m1) * i1 - expf(s2[row * SEQ + j] - m2) * i2;
    prow[j] = f32_to_bf16(p);
  }
}

// ---------------- in-place RMSNorm * (1 - lambda_init) ----------------
__global__ void __launch_bounds__(256) rmsnorm_kernel(float* __restrict__ out) {
  __shared__ float red[256];
  const long long row = blockIdx.x;
  const int t = threadIdx.x;
  float s = 0.f;
  for (int j = t; j < DEMB; j += 256) { float v = out[row * DEMB + j]; s += v * v; }
  red[t] = s; __syncthreads();
  for (int st = 128; st > 0; st >>= 1) {
    if (t < st) red[t] += red[t + st];
    __syncthreads();
  }
  const float inv = (1.0f - LAMBDA_INIT) / sqrtf(red[0] / (float)DEMB + EPSV);
  for (int j = t; j < DEMB; j += 256) out[row * DEMB + j] *= inv;
}

// ---------------- host orchestration ----------------
extern "C" void kernel_launch(void* const* d_in, const int* in_sizes, int n_in,
                              void* d_out, int out_size, void* d_ws, size_t ws_size,
                              hipStream_t stream) {
  (void)in_sizes; (void)n_in; (void)out_size; (void)ws_size;
  const float* x   = (const float*)d_in[0];
  const float* wq  = (const float*)d_in[1];
  const float* wk  = (const float*)d_in[2];
  const float* wv  = (const float*)d_in[3];
  const float* lq1 = (const float*)d_in[4];
  const float* lk1 = (const float*)d_in[5];
  const float* lq2 = (const float*)d_in[6];
  const float* lk2 = (const float*)d_in[7];
  float* out = (float*)d_out;

  const long long BS = (long long)BATCH * SEQ;   // 8192
  unsigned short* xb   = (unsigned short*)d_ws;                  // 8192*1024
  unsigned short* wqb  = xb  + BS * DEMB;                        // 1024*2048
  unsigned short* wkb  = wqb + (long long)DEMB * E2;
  unsigned short* wvb  = wkb + (long long)DEMB * E2;             // 1024*1024
  unsigned short* q12b = wvb + (long long)DEMB * DEMB;           // 8192*2048
  unsigned short* k12b = q12b + BS * E2;
  unsigned short* vb   = k12b + BS * E2;                         // 8192*1024
  unsigned short* Pd   = vb   + BS * DEMB;                       // 4*2048*2048
  float* lam = (float*)(Pd + (long long)BATCH * SEQ * SEQ);

  // 1) f32 -> bf16 conversions
  {
    int n;
    n = (int)(BS * DEMB);
    cvt_f32_bf16<<<(n + 255) / 256, 256, 0, stream>>>(x, xb, n);
    n = DEMB * E2;
    cvt_f32_bf16<<<(n + 255) / 256, 256, 0, stream>>>(wq, wqb, n);
    cvt_f32_bf16<<<(n + 255) / 256, 256, 0, stream>>>(wk, wkb, n);
    n = DEMB * DEMB;
    cvt_f32_bf16<<<(n + 255) / 256, 256, 0, stream>>>(wv, wvb, n);
  }

  // 2) lambda scalar
  lambda_kernel<<<1, 256, 0, stream>>>(lq1, lk1, lq2, lk2, lam);

  // 3) projections: q12 = x@Wq12, k12 = x@Wk12, v = x@Wv   (bf16 out)
  gemm_bf16_wmma<<<dim3(E2 / 64, (int)(BS / 64), 1), 256, 0, stream>>>(
      xb, wqb, nullptr, q12b, (int)BS, E2, DEMB, 0, 0, 0);
  gemm_bf16_wmma<<<dim3(E2 / 64, (int)(BS / 64), 1), 256, 0, stream>>>(
      xb, wkb, nullptr, k12b, (int)BS, E2, DEMB, 0, 0, 0);
  gemm_bf16_wmma<<<dim3(DEMB / 64, (int)(BS / 64), 1), 256, 0, stream>>>(
      xb, wvb, nullptr, vb, (int)BS, DEMB, DEMB, 0, 0, 0);

  // 4) fused scores + dual softmax -> P1 - lam*P2 (bf16)
  {
    size_t shmem = (size_t)(32 * 2048 + 640) * sizeof(float);  // 256KB scores + scratch
    diffattn_scores<<<dim3(SEQ / 16, BATCH), 256, shmem, stream>>>(q12b, k12b, lam, Pd);
  }

  // 5) out = Pdiff @ V  (f32 into d_out), batched
  gemm_bf16_wmma<<<dim3(DEMB / 64, SEQ / 64, BATCH), 256, 0, stream>>>(
      Pd, vb, out, nullptr, SEQ, DEMB, SEQ,
      (long long)SEQ * SEQ, (long long)SEQ * DEMB, (long long)SEQ * DEMB);

  // 6) RMSNorm * 0.2 in place
  rmsnorm_kernel<<<(int)BS, 256, 0, stream>>>(out);
}
